// PointNet2Cap_20529943675327
// MI455X (gfx1250) — matrix operand order repaired and checked by reading, not compiled, not run on hardware
//
#include <hip/hip_runtime.h>

typedef __attribute__((ext_vector_type(16))) _Float16 v16h;
typedef __attribute__((ext_vector_type(8)))  _Float16 v8h;
typedef __attribute__((ext_vector_type(8)))  float    v8f;

struct LayerDesc {
  const _Float16* w;    // (cout x cinPad) f16, BN-scale folded, zero-padded in K
  const float*    bias; // (cout) folded: g*s*b + beta
  int cinPad;
  int cout;
};

// ---------------------------------------------------------------------------
// small utility kernels
// ---------------------------------------------------------------------------
__global__ void extract_xyz_kernel(const float* __restrict__ pts,
                                   float* __restrict__ xyz, int total) {
  int i = blockIdx.x * blockDim.x + threadIdx.x;
  if (i < total) xyz[i] = pts[(i / 3) * 8 + (i % 3)];
}

__global__ void gather_centroids_kernel(const float* __restrict__ xyz,
                                        const int* __restrict__ fidx,
                                        float* __restrict__ nxyz,
                                        int n, int np, int total) {
  int i = blockIdx.x * blockDim.x + threadIdx.x;
  if (i >= total) return;
  int bs = i / 3, c = i - bs * 3;
  int b = bs / np;
  nxyz[i] = xyz[((size_t)b * n + fidx[bs]) * 3 + c];
}

__global__ void concat_g3_kernel(const float* __restrict__ x2,
                                 const float* __restrict__ l2,
                                 float* __restrict__ g, int total) {
  int i = blockIdx.x * blockDim.x + threadIdx.x;
  if (i >= total) return;
  int bs = i / 643, c = i - bs * 643;
  g[i] = (c < 3) ? x2[bs * 3 + c] : l2[(size_t)bs * 640 + (c - 3)];
}

// Fold BN into weights/bias, pad K to 32, convert to f16 in WMMA-B friendly
// (cout x cinPad) row-major layout (each lane fragment = contiguous 32B).
__global__ void prep_layer_kernel(const float* __restrict__ W, const float* __restrict__ b,
                                  const float* __restrict__ g, const float* __restrict__ be,
                                  _Float16* __restrict__ wOut, float* __restrict__ bOut,
                                  int cin, int cinPad, int cout) {
  int i = blockIdx.x * blockDim.x + threadIdx.x;
  int total = cout * cinPad;
  if (i >= total) return;
  const float s = rsqrtf(1.0f + 1e-5f);
  int o = i / cinPad, c = i - o * cinPad;
  float w = (c < cin) ? W[(size_t)o * cin + c] * g[o] * s : 0.0f;
  wOut[i] = (_Float16)w;
  if (c == 0) bOut[o] = g[o] * s * b[o] + be[o];
}

// ---------------------------------------------------------------------------
// farthest point sampling: one workgroup per batch, LDS dist + argmax reduce.
// Matches jax.lax.scan semantics: out[0]=0, out[it]=argmax after it updates.
// Tie-break = first (lowest) index, like jnp.argmax.
// ---------------------------------------------------------------------------
__global__ void __launch_bounds__(256)
fps_kernel(const float* __restrict__ xyz, int n, int npoint, int* __restrict__ outIdx) {
  __shared__ float dist[4096];
  __shared__ float rv[256];
  __shared__ int   ri[256];
  const int b = blockIdx.x, tid = threadIdx.x;
  const float* X = xyz + (size_t)b * n * 3;
  for (int j = tid; j < n; j += 256) dist[j] = 1e10f;
  if (tid == 0) outIdx[(size_t)b * npoint] = 0;
  __syncthreads();
  int far = 0;
  for (int it = 1; it < npoint; ++it) {
    float cx = X[far * 3], cy = X[far * 3 + 1], cz = X[far * 3 + 2];
    float bv = -1.0f; int bi = 0;
    for (int j = tid; j < n; j += 256) {
      float dx = X[j * 3] - cx, dy = X[j * 3 + 1] - cy, dz = X[j * 3 + 2] - cz;
      float d = dx * dx + dy * dy + dz * dz;
      float nd = fminf(dist[j], d);
      dist[j] = nd;
      if (nd > bv) { bv = nd; bi = j; }  // ascending j -> first max kept
    }
    rv[tid] = bv; ri[tid] = bi;
    __syncthreads();
    for (int off = 128; off > 0; off >>= 1) {
      if (tid < off) {
        float v2 = rv[tid + off]; int i2 = ri[tid + off];
        if (v2 > rv[tid] || (v2 == rv[tid] && i2 < ri[tid])) { rv[tid] = v2; ri[tid] = i2; }
      }
      __syncthreads();
    }
    far = ri[0];
    if (tid == 0) outIdx[(size_t)b * npoint + it] = far;
    __syncthreads();
  }
}

// ---------------------------------------------------------------------------
// query_ball: K smallest indices with d^2 <= r^2, padded with first hit.
// One thread per centroid (coalesced broadcast scan of xyz).
// ---------------------------------------------------------------------------
__global__ void query_ball_kernel(const float* __restrict__ xyz,
                                  const float* __restrict__ nxyz,
                                  int n, int np, int K, float r2, int total,
                                  int* __restrict__ idx) {
  int i = blockIdx.x * blockDim.x + threadIdx.x;
  if (i >= total) return;
  int b = i / np;
  float cx = nxyz[i * 3], cy = nxyz[i * 3 + 1], cz = nxyz[i * 3 + 2];
  const float* X = xyz + (size_t)b * n * 3;
  int cnt = 0, first = 0; bool haveFirst = false;
  for (int j = 0; j < n && cnt < K; ++j) {
    float dx = X[j * 3] - cx, dy = X[j * 3 + 1] - cy, dz = X[j * 3 + 2] - cz;
    float d = dx * dx + dy * dy + dz * dz;
    if (d <= r2) {
      if (!haveFirst) { first = j; haveFirst = true; }
      idx[(size_t)i * K + cnt] = j;
      ++cnt;
    }
  }
  for (; cnt < K; ++cnt) idx[(size_t)i * K + cnt] = first;
}

// ---------------------------------------------------------------------------
// 16x16 output-tile GEMM + bias + relu using v_wmma_f32_16x16x32_f16.
// A: LDS activations (rows x cinPad, f16). W: global (cout x cinPad, f16).
// A fragment: lane L row=L&15; K=k+e+16*hi (two 16B loads).
// B fragment: lane L col=L&15; K=k+e+16*hi  => one contiguous 32B load.
// C/D layout: VGPR r -> M = r + 8*hi, N = tileN*16 + (L&15).
// ---------------------------------------------------------------------------
__device__ __forceinline__ void gemm_bias_relu(
    const _Float16* __restrict__ A, int cinPad,
    const _Float16* __restrict__ W, const float* __restrict__ bias, int cout,
    _Float16* __restrict__ O, int oStride, int mTiles, int lane, int wave) {
  const int nTiles = cout >> 4;
  const int hi = (lane >= 16) ? 1 : 0;
  const int lid = lane & 15;
  for (int t = wave; t < mTiles * nTiles; t += 8) {
    const int tm = t / nTiles;
    const int tn = t - tm * nTiles;
    const _Float16* aPtr = A + (size_t)(tm * 16 + lid) * cinPad + hi * 16;
    const _Float16* bPtr = W + (size_t)(tn * 16 + lid) * cinPad + hi * 16;
    v8f acc = {0.f, 0.f, 0.f, 0.f, 0.f, 0.f, 0.f, 0.f};
    for (int k = 0; k < cinPad; k += 32) {
      v8h alo = *(const v8h*)(aPtr + k);
      v8h ahi = *(const v8h*)(aPtr + k + 8);
      v16h a = __builtin_shufflevector(alo, ahi,
                 0, 1, 2, 3, 4, 5, 6, 7, 8, 9, 10, 11, 12, 13, 14, 15);
      v16h bf = *(const v16h*)(bPtr + k);
      acc = __builtin_amdgcn_wmma_f32_16x16x32_f16(false, a, false, bf,
                                                   (short)0, acc, false, false);
    }
    const int nIdx = tn * 16 + lid;
    const float bb = bias[nIdx];
    const int rBase = tm * 16 + hi * 8;
#pragma unroll
    for (int r = 0; r < 8; ++r) {
      float v = acc[r] + bb;
      v = v > 0.f ? v : 0.f;
      O[(size_t)(rBase + r) * oStride + nIdx] = (_Float16)v;
    }
  }
}

// ---------------------------------------------------------------------------
// Fused grouped-MLP + max-pool branch kernel.
// One workgroup (8 wave32) per (batch, centroid). Streams ROWS-sample tiles
// through the 3-layer MLP in LDS with a running channel max.
// ---------------------------------------------------------------------------
template <int ROWS, int C0, int C1>
__global__ void __launch_bounds__(256)
mlp_branch_kernel(const float* __restrict__ feats, int n, int Cfeat,
                  const float* __restrict__ xyz, const float* __restrict__ nxyz,
                  const int* __restrict__ idx,
                  int np, int K, int Cxyz,
                  LayerDesc L0, LayerDesc L1, LayerDesc L2,
                  float* __restrict__ out, int outStride, int chOff) {
  __shared__ __align__(32) _Float16 bufA[ROWS * C0];
  __shared__ __align__(32) _Float16 bufB[ROWS * C1];
  __shared__ float maxbuf[C1];

  const int s = blockIdx.x, b = blockIdx.y;
  const int tid = threadIdx.x, lane = tid & 31, wave = tid >> 5;
  const int base = b * np + s;
  const int cin0 = Cfeat + Cxyz;
  const int cinPad0 = L0.cinPad;
  const int mTiles = ROWS / 16;

  for (int c = tid; c < C1; c += 256) maxbuf[c] = 0.f;  // relu outputs >= 0

  for (int t0 = 0; t0 < K; t0 += ROWS) {
    __syncthreads();
    // stage gathered input: [feats(Cfeat) | xyz - centroid (Cxyz) | zero-pad]
    const int total0 = ROWS * cinPad0;
    for (int f = tid; f < total0; f += 256) {
      const int j = f / cinPad0, c = f - j * cinPad0;
      const int r = t0 + j;
      const int gid = idx ? idx[(size_t)base * K + r] : r;
      float v = 0.f;
      if (c < Cfeat) {
        v = feats[((size_t)b * n + gid) * Cfeat + c];
      } else if (c < cin0) {
        const int cc = c - Cfeat;
        v = xyz[((size_t)b * n + gid) * 3 + cc] - nxyz[(size_t)base * 3 + cc];
      }
      bufA[f] = (_Float16)v;
    }
    __syncthreads();
    gemm_bias_relu(bufA, L0.cinPad, L0.w, L0.bias, L0.cout, bufB, L0.cout, mTiles, lane, wave);
    __syncthreads();
    gemm_bias_relu(bufB, L1.cinPad, L1.w, L1.bias, L1.cout, bufA, L1.cout, mTiles, lane, wave);
    __syncthreads();
    gemm_bias_relu(bufA, L2.cinPad, L2.w, L2.bias, L2.cout, bufB, L2.cout, mTiles, lane, wave);
    __syncthreads();
    for (int c = tid; c < L2.cout; c += 256) {
      float m = maxbuf[c];
      for (int j = 0; j < ROWS; ++j)
        m = fmaxf(m, (float)bufB[(size_t)j * L2.cout + c]);
      maxbuf[c] = m;
    }
  }
  __syncthreads();
  for (int c = tid; c < L2.cout; c += 256)
    out[(size_t)base * outStride + chOff + c] = maxbuf[c];
}

// ---------------------------------------------------------------------------
// Head: x(16x1024) -> 512 -> 256 -> 64 (BN+relu) -> 1.  Single workgroup,
// one 16-row WMMA tile chain, final 64->1 as scalar dot in f32.
// ---------------------------------------------------------------------------
__global__ void __launch_bounds__(256)
head_kernel(const float* __restrict__ x,
            LayerDesc H0, LayerDesc H1, LayerDesc H2,
            const float* __restrict__ wf, const float* __restrict__ bf,
            float* __restrict__ out) {
  __shared__ __align__(32) _Float16 bufA[16 * 1024];
  __shared__ __align__(32) _Float16 bufB[16 * 512];
  const int tid = threadIdx.x, lane = tid & 31, wave = tid >> 5;
  for (int f = tid; f < 16 * 1024; f += 256) bufA[f] = (_Float16)x[f];
  __syncthreads();
  gemm_bias_relu(bufA, 1024, H0.w, H0.bias, 512, bufB, 512, 1, lane, wave);
  __syncthreads();
  gemm_bias_relu(bufB, 512, H1.w, H1.bias, 256, bufA, 256, 1, lane, wave);
  __syncthreads();
  gemm_bias_relu(bufA, 256, H2.w, H2.bias, 64, bufB, 64, 1, lane, wave);
  __syncthreads();
  if (tid < 16) {
    float sum = bf[0];
    for (int c = 0; c < 64; ++c) sum += (float)bufB[tid * 64 + c] * wf[c];
    out[tid] = sum;
  }
}

// ---------------------------------------------------------------------------
// host orchestration
// ---------------------------------------------------------------------------
extern "C" void kernel_launch(void* const* d_in, const int* in_sizes, int n_in,
                              void* d_out, int out_size, void* d_ws, size_t ws_size,
                              hipStream_t stream) {
  (void)in_sizes; (void)n_in; (void)out_size; (void)ws_size;
  const int B = 16, N = 4096;
  auto inF = [&](int i) { return (const float*)d_in[i]; };
  const float* points = inF(0);

  char* wsP = (char*)d_ws;
  auto alloc = [&](size_t bytes) -> void* {
    void* p = (void*)wsP;
    wsP += (bytes + 255) & ~(size_t)255;
    return p;
  };

  // ---- prep all BN-folded f16 weights ----
  auto prep = [&](int baseIdx, int t, int cin, int cout) -> LayerDesc {
    int cinPad = (cin + 31) & ~31;
    _Float16* w = (_Float16*)alloc((size_t)cout * cinPad * sizeof(_Float16));
    float* bias = (float*)alloc((size_t)cout * sizeof(float));
    int total = cout * cinPad;
    prep_layer_kernel<<<(total + 255) / 256, 256, 0, stream>>>(
        inF(baseIdx + 4 * t), inF(baseIdx + 4 * t + 1),
        inF(baseIdx + 4 * t + 2), inF(baseIdx + 4 * t + 3),
        w, bias, cin, cinPad, cout);
    return LayerDesc{w, bias, cinPad, cout};
  };

  const int sa1ci[9] = {11, 32, 32, 11, 64, 64, 11, 64, 96};
  const int sa1co[9] = {32, 32, 64, 64, 64, 128, 64, 96, 128};
  const int sa2ci[9] = {323, 64, 64, 323, 128, 128, 323, 128, 128};
  const int sa2co[9] = {64, 64, 128, 128, 128, 256, 128, 128, 256};
  const int sa3ci[3] = {643, 256, 512};
  const int sa3co[3] = {256, 512, 1024};
  const int hci[3] = {1024, 512, 256};
  const int hco[3] = {512, 256, 64};

  LayerDesc sa1[9], sa2[9], sa3[3], head[3];
  for (int t = 0; t < 9; ++t) sa1[t] = prep(1, t, sa1ci[t], sa1co[t]);
  for (int t = 0; t < 9; ++t) sa2[t] = prep(37, t, sa2ci[t], sa2co[t]);
  for (int t = 0; t < 3; ++t) sa3[t] = prep(73, t, sa3ci[t], sa3co[t]);
  for (int t = 0; t < 3; ++t) head[t] = prep(85, t, hci[t], hco[t]);
  const float* wf = inF(97);  // (1,64)
  const float* bf = inF(98);  // (1,)

  // ---- activation scratch ----
  float* xyz1 = (float*)alloc((size_t)B * N * 3 * 4);
  int*   fps1 = (int*)alloc((size_t)B * 512 * 4);
  float* nx1  = (float*)alloc((size_t)B * 512 * 3 * 4);
  int* idxA = (int*)alloc((size_t)B * 512 * 16 * 4);
  int* idxB = (int*)alloc((size_t)B * 512 * 32 * 4);
  int* idxC = (int*)alloc((size_t)B * 512 * 128 * 4);
  float* l1 = (float*)alloc((size_t)B * 512 * 320 * 4);
  int*   fps2 = (int*)alloc((size_t)B * 128 * 4);
  float* nx2  = (float*)alloc((size_t)B * 128 * 3 * 4);
  int* idxD = (int*)alloc((size_t)B * 128 * 32 * 4);
  int* idxE = (int*)alloc((size_t)B * 128 * 64 * 4);
  int* idxF = (int*)alloc((size_t)B * 128 * 128 * 4);
  float* l2 = (float*)alloc((size_t)B * 128 * 640 * 4);
  float* g3 = (float*)alloc((size_t)B * 128 * 643 * 4);
  float* s3o = (float*)alloc((size_t)B * 1024 * 4);

  // ---- stage 1: SA1 ----
  {
    int tot = B * N * 3;
    extract_xyz_kernel<<<(tot + 255) / 256, 256, 0, stream>>>(points, xyz1, tot);
  }
  fps_kernel<<<B, 256, 0, stream>>>(xyz1, N, 512, fps1);
  {
    int tot = B * 512 * 3;
    gather_centroids_kernel<<<(tot + 255) / 256, 256, 0, stream>>>(xyz1, fps1, nx1, N, 512, tot);
  }
  {
    int tot = B * 512;
    query_ball_kernel<<<(tot + 255) / 256, 256, 0, stream>>>(xyz1, nx1, N, 512, 16, 0.01f, tot, idxA);
    query_ball_kernel<<<(tot + 255) / 256, 256, 0, stream>>>(xyz1, nx1, N, 512, 32, 0.04f, tot, idxB);
    query_ball_kernel<<<(tot + 255) / 256, 256, 0, stream>>>(xyz1, nx1, N, 512, 128, 0.16f, tot, idxC);
  }
  mlp_branch_kernel<16, 32, 64><<<dim3(512, B), 256, 0, stream>>>(
      points, N, 8, xyz1, nx1, idxA, 512, 16, 3, sa1[0], sa1[1], sa1[2], l1, 320, 0);
  mlp_branch_kernel<32, 64, 128><<<dim3(512, B), 256, 0, stream>>>(
      points, N, 8, xyz1, nx1, idxB, 512, 32, 3, sa1[3], sa1[4], sa1[5], l1, 320, 64);
  mlp_branch_kernel<32, 96, 128><<<dim3(512, B), 256, 0, stream>>>(
      points, N, 8, xyz1, nx1, idxC, 512, 128, 3, sa1[6], sa1[7], sa1[8], l1, 320, 192);

  // ---- stage 2: SA2 ----
  fps_kernel<<<B, 256, 0, stream>>>(nx1, 512, 128, fps2);
  {
    int tot = B * 128 * 3;
    gather_centroids_kernel<<<(tot + 255) / 256, 256, 0, stream>>>(nx1, fps2, nx2, 512, 128, tot);
  }
  {
    int tot = B * 128;
    query_ball_kernel<<<(tot + 255) / 256, 256, 0, stream>>>(nx1, nx2, 512, 128, 32, 0.04f, tot, idxD);
    query_ball_kernel<<<(tot + 255) / 256, 256, 0, stream>>>(nx1, nx2, 512, 128, 64, 0.16f, tot, idxE);
    query_ball_kernel<<<(tot + 255) / 256, 256, 0, stream>>>(nx1, nx2, 512, 128, 128, 0.64f, tot, idxF);
  }
  mlp_branch_kernel<32, 352, 128><<<dim3(128, B), 256, 0, stream>>>(
      l1, 512, 320, nx1, nx2, idxD, 128, 32, 3, sa2[0], sa2[1], sa2[2], l2, 640, 0);
  mlp_branch_kernel<32, 352, 256><<<dim3(128, B), 256, 0, stream>>>(
      l1, 512, 320, nx1, nx2, idxE, 128, 64, 3, sa2[3], sa2[4], sa2[5], l2, 640, 128);
  mlp_branch_kernel<32, 352, 256><<<dim3(128, B), 256, 0, stream>>>(
      l1, 512, 320, nx1, nx2, idxF, 128, 128, 3, sa2[6], sa2[7], sa2[8], l2, 640, 384);

  // ---- stage 3: SA3 (group-all) ----
  {
    int tot = B * 128 * 643;
    concat_g3_kernel<<<(tot + 255) / 256, 256, 0, stream>>>(nx2, l2, g3, tot);
  }
  mlp_branch_kernel<16, 672, 1024><<<dim3(1, B), 256, 0, stream>>>(
      g3, 128, 643, nullptr, nullptr, nullptr, 1, 128, 0,
      sa3[0], sa3[1], sa3[2], s3o, 1024, 0);

  // ---- head ----
  head_kernel<<<1, 256, 0, stream>>>(s3o, head[0], head[1], head[2], wf, bf, (float*)d_out);
}
